// RoleAllocation_19533511262236
// MI455X (gfx1250) — matrix operand order repaired and checked by reading, not compiled, not run on hardware
//
#include <hip/hip_runtime.h>
#include <hip/hip_bf16.h>
#include <math.h>

// ---------------- problem constants ----------------
#define BB    32
#define NR    4096
#define D_IN  384
#define D_CTX 128
#define HH    64
#define AA    16
#define ROWS  (BB*NR)            // 131072
#define LOG_VAR2f (-4.6051701860f)   // log(0.01)
#define INV_VAR2f (100.0f)

typedef __attribute__((ext_vector_type(16))) _Float16 v16h;
typedef __attribute__((ext_vector_type(8)))  float    v8f;

union HV { v16h v; unsigned u[8]; _Float16 h[16]; };

static __device__ __forceinline__ v8f wmma16(v16h a, v16h b, v8f c) {
  // D = A(16x32 f16) x B(32x16 f16) + C(16x16 f32)
  return __builtin_amdgcn_wmma_f32_16x16x32_f16(
      /*neg_a=*/false, a, /*neg_b=*/false, b,
      /*c_mod=*/(short)0, c, /*reuse_a=*/false, /*reuse_b=*/false);
}

// ---------------------------------------------------------------
// Kernel A: repack a KxN f32 weight into WMMA B-fragment layout.
// B-matrix (32x16 f16) per ISA 7.12.2: lane holds column n=lane&15,
// lanes 0-15 hold K 0..7,16..23 (2 per dword), lanes 16-31 hold K 8..15,24..31.
// Fragment f = kt*NT + nt; dword idx = (f*32 + lane)*8 + i.
// ---------------------------------------------------------------
__global__ void prep_frag(const float* __restrict__ W, unsigned* __restrict__ F,
                          int K, int N) {
  const int ntc  = N >> 4;
  const int frag = blockIdx.x;          // kt*ntc + nt
  const int kt   = frag / ntc;
  const int nt   = frag % ntc;
  const int lane = threadIdx.x >> 3;    // 0..31
  const int i    = threadIdx.x & 7;     // 0..7
  const int n    = nt * 16 + (lane & 15);
  const int k0   = kt * 32 + ((i >> 2) << 4) + ((lane >> 4) << 3) + ((i & 3) << 1);
  union { _Float16 h[2]; unsigned u; } p;
  p.h[0] = (_Float16)W[(size_t)k0 * N + n];
  p.h[1] = (_Float16)W[(size_t)(k0 + 1) * N + n];
  F[((size_t)frag * 32 + lane) * 8 + i] = p.u;
}

// ---------------------------------------------------------------
// Kernel B: fused VAE. One wave = one 16-row tile. 8 waves/block.
// Dynamic LDS: fbuf[12288] u32 (W1 frags, later reused for W4 frags)
//              sbuf[6144]  u32 (W21 | W22 | W3 frags)
//              tbuf[8192]  f16 (8 waves x 16x64 transpose staging)
// ---------------------------------------------------------------
__global__ __launch_bounds__(256) void vae_kernel(
    const float* __restrict__ roles, const float* __restrict__ eps,
    const float* __restrict__ b1,  const float* __restrict__ b21,
    const float* __restrict__ b22, const float* __restrict__ b3,
    const float* __restrict__ b4,
    const unsigned* __restrict__ w1f, const unsigned* __restrict__ w21f,
    const unsigned* __restrict__ w22f, const unsigned* __restrict__ w3f,
    const unsigned* __restrict__ w4f,
    float* __restrict__ remb, float* __restrict__ part) {
  extern __shared__ unsigned dsm[];
  unsigned* fbuf = dsm;                 // 12288 dwords (48KB)
  unsigned* sbuf = dsm + 12288;         //  6144 dwords (24KB)
  _Float16* tbuf = (_Float16*)(dsm + 12288 + 6144);  // 8192 halves (16KB)

  const int tid  = threadIdx.x;
  const int w    = tid >> 5;
  const int lane = tid & 31;
  const int hl   = lane >> 4;           // half-wave id
  const int ln   = lane & 15;

  // stage W1 + small weight fragments into LDS
  for (int idx = tid; idx < 12288; idx += 256) fbuf[idx] = w1f[idx];
  for (int idx = tid; idx < 2048; idx += 256) {
    sbuf[idx]        = w21f[idx];
    sbuf[2048 + idx] = w22f[idx];
    sbuf[4096 + idx] = w3f[idx];
  }
  __syncthreads();

  const int rowbase = blockIdx.x * 128 + w * 16;

  // ---------------- h = relu(roles @ W1 + b1) ----------------
  v8f acc[4] = {{}, {}, {}, {}};
  const float* arp = roles + (size_t)(rowbase + ln) * D_IN + hl * 8;
  for (int kt = 0; kt < 12; ++kt) {
    const float* p = arp + kt * 32;
    float4 f0 = *(const float4*)(p);
    float4 f1 = *(const float4*)(p + 4);
    float4 f2 = *(const float4*)(p + 16);
    float4 f3 = *(const float4*)(p + 20);
    HV a;
    a.h[0]=(_Float16)f0.x;  a.h[1]=(_Float16)f0.y;  a.h[2]=(_Float16)f0.z;  a.h[3]=(_Float16)f0.w;
    a.h[4]=(_Float16)f1.x;  a.h[5]=(_Float16)f1.y;  a.h[6]=(_Float16)f1.z;  a.h[7]=(_Float16)f1.w;
    a.h[8]=(_Float16)f2.x;  a.h[9]=(_Float16)f2.y;  a.h[10]=(_Float16)f2.z; a.h[11]=(_Float16)f2.w;
    a.h[12]=(_Float16)f3.x; a.h[13]=(_Float16)f3.y; a.h[14]=(_Float16)f3.z; a.h[15]=(_Float16)f3.w;
#pragma unroll
    for (int nt = 0; nt < 4; ++nt) {
      const unsigned* bp = fbuf + ((size_t)((kt * 4 + nt) * 32 + lane)) * 8;
      HV bf;
      uint4 q0 = *(const uint4*)bp;
      uint4 q1 = *(const uint4*)(bp + 4);
      bf.u[0]=q0.x; bf.u[1]=q0.y; bf.u[2]=q0.z; bf.u[3]=q0.w;
      bf.u[4]=q1.x; bf.u[5]=q1.y; bf.u[6]=q1.z; bf.u[7]=q1.w;
      acc[nt] = wmma16(a.v, bf.v, acc[nt]);
    }
  }

  // store h (relu + bias) into per-wave LDS tile, row-major 16x64 f16
  _Float16* tw = tbuf + w * 1024;
  const unsigned* twu = (const unsigned*)tw;
  {
    float bb[4];
#pragma unroll
    for (int nt = 0; nt < 4; ++nt) bb[nt] = b1[nt * 16 + ln];
#pragma unroll
    for (int nt = 0; nt < 4; ++nt)
#pragma unroll
      for (int r = 0; r < 8; ++r) {
        float hv = fmaxf(acc[nt][r] + bb[nt], 0.0f);
        int m = r + hl * 8;
        tw[m * 64 + nt * 16 + ln] = (_Float16)hv;
      }
  }
  __syncthreads();

  // load h as two A-fragments (16x32 each)
  HV ha[2];
#pragma unroll
  for (int kt = 0; kt < 2; ++kt)
#pragma unroll
    for (int i = 0; i < 8; ++i) {
      int k0 = kt * 32 + ((i >> 2) << 4) + hl * 8 + ((i & 3) << 1);
      ha[kt].u[i] = twu[(ln * 64 + k0) >> 1];
    }

  // ---------------- mu / log_var ----------------
  v8f ma[4] = {{}, {}, {}, {}};
  v8f la[4] = {{}, {}, {}, {}};
#pragma unroll
  for (int kt = 0; kt < 2; ++kt)
#pragma unroll
    for (int nt = 0; nt < 4; ++nt) {
      const unsigned* bp1 = sbuf + ((size_t)((kt * 4 + nt) * 32 + lane)) * 8;
      const unsigned* bp2 = bp1 + 2048;
      HV bf1, bf2;
      uint4 q0 = *(const uint4*)bp1; uint4 q1 = *(const uint4*)(bp1 + 4);
      bf1.u[0]=q0.x; bf1.u[1]=q0.y; bf1.u[2]=q0.z; bf1.u[3]=q0.w;
      bf1.u[4]=q1.x; bf1.u[5]=q1.y; bf1.u[6]=q1.z; bf1.u[7]=q1.w;
      uint4 r0 = *(const uint4*)bp2; uint4 r1 = *(const uint4*)(bp2 + 4);
      bf2.u[0]=r0.x; bf2.u[1]=r0.y; bf2.u[2]=r0.z; bf2.u[3]=r0.w;
      bf2.u[4]=r1.x; bf2.u[5]=r1.y; bf2.u[6]=r1.z; bf2.u[7]=r1.w;
      ma[nt] = wmma16(ha[kt].v, bf1.v, ma[nt]);
      la[nt] = wmma16(ha[kt].v, bf2.v, la[nt]);
    }

  // ---------------- z, KLD, l2norm(z) -> role_emb ----------------
  float kacc = 0.0f;
  float zval[4][8];
  float srq[8] = {0,0,0,0,0,0,0,0};
  {
    float bm[4], bl[4];
#pragma unroll
    for (int nt = 0; nt < 4; ++nt) { bm[nt] = b21[nt*16+ln]; bl[nt] = b22[nt*16+ln]; }
#pragma unroll
    for (int nt = 0; nt < 4; ++nt)
#pragma unroll
      for (int r = 0; r < 8; ++r) {
        float mu = ma[nt][r] + bm[nt];
        float lv = la[nt][r] + bl[nt];
        int m = r + hl * 8;
        float ev = eps[(size_t)(rowbase + m) * HH + nt * 16 + ln];
        float z  = mu + ev * (expf(0.5f * lv) * 0.1f);
        kacc += 1.0f - LOG_VAR2f + lv - (mu * mu + expf(lv)) * INV_VAR2f;
        zval[nt][r] = z;
        srq[r] += z * z;
      }
  }
#pragma unroll
  for (int r = 0; r < 8; ++r) {
    float s = srq[r];
    s += __shfl_xor(s, 1, 32);
    s += __shfl_xor(s, 2, 32);
    s += __shfl_xor(s, 4, 32);
    s += __shfl_xor(s, 8, 32);
    srq[r] = 1.0f / fmaxf(sqrtf(s), 1e-12f);
  }
#pragma unroll
  for (int nt = 0; nt < 4; ++nt)
#pragma unroll
    for (int r = 0; r < 8; ++r) {
      int m = r + hl * 8;
      remb[(size_t)(rowbase + m) * HH + nt * 16 + ln] = zval[nt][r] * srq[r];
    }

  __syncthreads();
  // store z tile for transpose; meanwhile swap fbuf to W4 fragments
#pragma unroll
  for (int nt = 0; nt < 4; ++nt)
#pragma unroll
    for (int r = 0; r < 8; ++r) {
      int m = r + hl * 8;
      tw[m * 64 + nt * 16 + ln] = (_Float16)zval[nt][r];
    }
  for (int idx = tid; idx < 12288; idx += 256) fbuf[idx] = w4f[idx];
  __syncthreads();

  HV za[2];
#pragma unroll
  for (int kt = 0; kt < 2; ++kt)
#pragma unroll
    for (int i = 0; i < 8; ++i) {
      int k0 = kt * 32 + ((i >> 2) << 4) + hl * 8 + ((i & 3) << 1);
      za[kt].u[i] = twu[(ln * 64 + k0) >> 1];
    }

  // ---------------- t = relu(z @ W3 + b3) ----------------
  v8f ta[4] = {{}, {}, {}, {}};
#pragma unroll
  for (int kt = 0; kt < 2; ++kt)
#pragma unroll
    for (int nt = 0; nt < 4; ++nt) {
      const unsigned* bp = sbuf + 4096 + ((size_t)((kt * 4 + nt) * 32 + lane)) * 8;
      HV bf;
      uint4 q0 = *(const uint4*)bp; uint4 q1 = *(const uint4*)(bp + 4);
      bf.u[0]=q0.x; bf.u[1]=q0.y; bf.u[2]=q0.z; bf.u[3]=q0.w;
      bf.u[4]=q1.x; bf.u[5]=q1.y; bf.u[6]=q1.z; bf.u[7]=q1.w;
      ta[nt] = wmma16(za[kt].v, bf.v, ta[nt]);
    }
  __syncthreads();
  {
    float bb[4];
#pragma unroll
    for (int nt = 0; nt < 4; ++nt) bb[nt] = b3[nt * 16 + ln];
#pragma unroll
    for (int nt = 0; nt < 4; ++nt)
#pragma unroll
      for (int r = 0; r < 8; ++r) {
        float tv = fmaxf(ta[nt][r] + bb[nt], 0.0f);
        int m = r + hl * 8;
        tw[m * 64 + nt * 16 + ln] = (_Float16)tv;
      }
  }
  __syncthreads();

  HV tA[2];
#pragma unroll
  for (int kt = 0; kt < 2; ++kt)
#pragma unroll
    for (int i = 0; i < 8; ++i) {
      int k0 = kt * 32 + ((i >> 2) << 4) + hl * 8 + ((i & 3) << 1);
      tA[kt].u[i] = twu[(ln * 64 + k0) >> 1];
    }

  // ---------------- x_hat = t @ W4 + b4; fused MSE ----------------
  float msacc = 0.0f;
  for (int nt = 0; nt < 24; ++nt) {
    v8f x = {};
#pragma unroll
    for (int kt = 0; kt < 2; ++kt) {
      const unsigned* bp = fbuf + ((size_t)((kt * 24 + nt) * 32 + lane)) * 8;
      HV bf;
      uint4 q0 = *(const uint4*)bp; uint4 q1 = *(const uint4*)(bp + 4);
      bf.u[0]=q0.x; bf.u[1]=q0.y; bf.u[2]=q0.z; bf.u[3]=q0.w;
      bf.u[4]=q1.x; bf.u[5]=q1.y; bf.u[6]=q1.z; bf.u[7]=q1.w;
      x = wmma16(tA[kt].v, bf.v, x);
    }
    float bb4 = b4[nt * 16 + ln];
#pragma unroll
    for (int r = 0; r < 8; ++r) {
      int m = r + hl * 8;
      float xv = x[r] + bb4;
      float rv = roles[(size_t)(rowbase + m) * D_IN + nt * 16 + ln];
      float d = xv - rv;
      msacc += d * d;
    }
  }

  // deterministic per-block reduction of mse / kld partials
  __syncthreads();
  float* red = (float*)tbuf;   // 4096 floats of scratch available
  red[tid] = msacc;
  red[256 + tid] = kacc;
  __syncthreads();
  for (int s = 128; s > 0; s >>= 1) {
    if (tid < s) { red[tid] += red[tid + s]; red[256 + tid] += red[256 + tid + s]; }
    __syncthreads();
  }
  if (tid == 0) {
    part[blockIdx.x * 2]     = red[0];
    part[blockIdx.x * 2 + 1] = red[256];
  }
}

// ---------------------------------------------------------------
// Kernel: deterministic final reduction -> vae_loss at out[2592]
// ---------------------------------------------------------------
__global__ void finalize_kernel(const float* __restrict__ part, float* __restrict__ out) {
  __shared__ float red[512];
  int t = threadIdx.x;
  float ms = 0.0f, kk = 0.0f;
  for (int i = t; i < 1024; i += 256) { ms += part[2 * i]; kk += part[2 * i + 1]; }
  red[t] = ms; red[256 + t] = kk;
  __syncthreads();
  for (int s = 128; s > 0; s >>= 1) {
    if (t < s) { red[t] += red[t + s]; red[256 + t] += red[256 + t + s]; }
    __syncthreads();
  }
  if (t == 0) {
    float mse_sum = red[0]   / ((float)NR * (float)D_IN);      // sum_b mse_b
    float kld_sum = -0.5f * red[256] / ((float)NR * (float)HH); // sum_b kld_b
    out[2592] = (mse_sum + kld_sum) / (float)BB;
  }
}

// ---------------------------------------------------------------
// Kernel C: sequential role selection, one block per batch item.
// ---------------------------------------------------------------
__global__ __launch_bounds__(256) void select_kernel(
    const float* __restrict__ remb, const float* __restrict__ contexts,
    const float* __restrict__ rand_vals, const float* __restrict__ Wc,
    const float* __restrict__ bc, const float* __restrict__ init_emb,
    float* __restrict__ out) {
  __shared__ float dots[NR];
  __shared__ float ctxh[D_CTX + HH];
  __shared__ float cvec[HH];
  __shared__ float hist[HH];
  __shared__ float cur[HH];
  __shared__ float part[256];
  __shared__ float red[256];
  __shared__ int ssel;

  const int b = blockIdx.x, t = threadIdx.x;
  if (t < D_CTX) ctxh[t] = contexts[b * D_CTX + t];
  if (t < HH) { hist[t] = init_emb[t]; cur[t] = init_emb[t]; }
  float lp = 0.0f;
  __syncthreads();

  const float* eb = remb + ((size_t)b << 18);   // b * 4096 * 64

  for (int step = 0; step < AA; ++step) {
    // ---- hist = layernorm(hist + cur) ----
    float v = (t < HH) ? (hist[t] + cur[t]) : 0.0f;
    red[t] = v;
    __syncthreads();
    for (int s = 128; s > 0; s >>= 1) { if (t < s) red[t] += red[t + s]; __syncthreads(); }
    float mean = red[0] * (1.0f / HH);
    __syncthreads();
    red[t] = (t < HH) ? (v - mean) * (v - mean) : 0.0f;
    __syncthreads();
    for (int s = 128; s > 0; s >>= 1) { if (t < s) red[t] += red[t + s]; __syncthreads(); }
    float var = red[0] * (1.0f / HH);
    __syncthreads();
    if (t < HH) {
      float hn = (v - mean) * rsqrtf(var + 1e-5f);
      hist[t] = hn;
      ctxh[D_CTX + t] = hn;
    }
    __syncthreads();

    // ---- c = normalize(concat(ctx,hist) @ Wc + bc) ----
    if (t < HH) {
      float a = bc[t];
      for (int i = 0; i < D_CTX + HH; ++i) a += ctxh[i] * Wc[i * HH + t];
      cvec[t] = a;
    }
    __syncthreads();
    red[t] = (t < HH) ? cvec[t] * cvec[t] : 0.0f;
    __syncthreads();
    for (int s = 128; s > 0; s >>= 1) { if (t < s) red[t] += red[t + s]; __syncthreads(); }
    float cinv = 1.0f / fmaxf(sqrtf(red[0]), 1e-12f);
    __syncthreads();
    if (t < HH) cvec[t] *= cinv;
    __syncthreads();

    // ---- scores = emb @ c ; contiguous 16-role chunk per thread ----
    float lmax = -3.4e38f;
    for (int rr = 0; rr < 16; ++rr) {
      int i = t * 16 + rr;
      const float4* ep = (const float4*)(eb + ((size_t)i << 6));
      float d = 0.0f;
#pragma unroll
      for (int q = 0; q < 16; ++q) {
        float4 e = ep[q];
        d += e.x * cvec[4*q] + e.y * cvec[4*q+1] + e.z * cvec[4*q+2] + e.w * cvec[4*q+3];
      }
      dots[i] = d;
      lmax = fmaxf(lmax, d);
    }
    red[t] = lmax;
    __syncthreads();
    for (int s = 128; s > 0; s >>= 1) { if (t < s) red[t] = fmaxf(red[t], red[t + s]); __syncthreads(); }
    float M = red[0];
    __syncthreads();

    float p = 0.0f;
    for (int rr = 0; rr < 16; ++rr) p += expf(dots[t * 16 + rr] - M);
    part[t] = p;
    red[t] = p;
    __syncthreads();
    for (int s = 128; s > 0; s >>= 1) { if (t < s) red[t] += red[t + s]; __syncthreads(); }
    float S = red[0];
    __syncthreads();

    // ---- inverse-CDF sample: first idx with inclusive cumsum > r*S ----
    if (t == 0) {
      float thr = rand_vals[b * AA + step] * S;
      float run = 0.0f;
      int seg = 255;
      for (int q = 0; q < 256; ++q) {
        if (run + part[q] > thr) { seg = q; break; }
        run += part[q];
      }
      int sel = seg * 16 + 15;
      for (int j = 0; j < 16; ++j) {
        run += expf(dots[seg * 16 + j] - M);
        if (run > thr) { sel = seg * 16 + j; break; }
      }
      lp += (dots[sel] - M) - logf(S);
      ssel = sel;
      out[b * AA + step] = (float)sel;
    }
    __syncthreads();
    int sel = ssel;
    if (t < HH) cur[t] = eb[((size_t)sel << 6) + t];
    __syncthreads();
  }

  if (t == 0) out[512 + b] = lp;
  if (t < HH) out[544 + b * HH + t] = hist[t];
}

// ---------------------------------------------------------------
extern "C" void kernel_launch(void* const* d_in, const int* in_sizes, int n_in,
                              void* d_out, int out_size, void* d_ws, size_t ws_size,
                              hipStream_t stream) {
  (void)in_sizes; (void)n_in; (void)out_size; (void)ws_size;
  const float* roles     = (const float*)d_in[0];
  const float* contexts  = (const float*)d_in[1];
  const float* eps       = (const float*)d_in[2];
  const float* rand_vals = (const float*)d_in[3];
  // d_in[4] = agent_num (constant 16, unused)
  const float* W1  = (const float*)d_in[5];
  const float* b1  = (const float*)d_in[6];
  const float* W21 = (const float*)d_in[7];
  const float* b21 = (const float*)d_in[8];
  const float* W22 = (const float*)d_in[9];
  const float* b22 = (const float*)d_in[10];
  const float* W3  = (const float*)d_in[11];
  const float* b3  = (const float*)d_in[12];
  const float* W4  = (const float*)d_in[13];
  const float* b4  = (const float*)d_in[14];
  const float* Wc  = (const float*)d_in[15];
  const float* bc  = (const float*)d_in[16];
  const float* init_emb = (const float*)d_in[17];
  float* out = (float*)d_out;

  // workspace layout
  char* ws = (char*)d_ws;
  float*    remb = (float*)ws;                       // 131072*64*4 = 32 MB
  unsigned* w1f  = (unsigned*)(ws + (size_t)ROWS * HH * 4);
  unsigned* w21f = w1f + 12288;
  unsigned* w22f = w21f + 2048;
  unsigned* w3f  = w22f + 2048;
  unsigned* w4f  = w3f + 2048;
  float*    part = (float*)(w4f + 12288);            // 1024*2 floats

  prep_frag<<<48, 256, 0, stream>>>(W1,  w1f,  D_IN, HH);
  prep_frag<<<8,  256, 0, stream>>>(W21, w21f, HH,   HH);
  prep_frag<<<8,  256, 0, stream>>>(W22, w22f, HH,   HH);
  prep_frag<<<8,  256, 0, stream>>>(W3,  w3f,  HH,   HH);
  prep_frag<<<48, 256, 0, stream>>>(W4,  w4f,  HH,   D_IN);

  // dynamic LDS: (12288 + 6144)*4 + 8192*2 = 90112 bytes
  vae_kernel<<<ROWS / 128, 256, 90112, stream>>>(
      roles, eps, b1, b21, b22, b3, b4,
      w1f, w21f, w22f, w3f, w4f, remb, part);

  finalize_kernel<<<1, 256, 0, stream>>>(part, out);

  select_kernel<<<BB, 256, 0, stream>>>(remb, contexts, rand_vals, Wc, bc,
                                        init_emb, out);
}
